// UltimateFusionV3_37323265802895
// MI455X (gfx1250) — matrix operand order repaired and checked by reading, not compiled, not run on hardware
//
#include <hip/hip_runtime.h>
#include <hip/hip_bf16.h>
#include <math.h>

#define BB   8
#define SS   512
#define DD   1024
#define NBLK 8
#define KSEL 4
#define MM   (BB*SS)                 /* 4096 rows */
#define BSD  ((size_t)BB*SS*DD)      /* 4,194,304 elements */

typedef __attribute__((ext_vector_type(16))) __bf16 v16bf;
typedef __attribute__((ext_vector_type(8)))  float  v8f;

__device__ __forceinline__ float sigf(float x) { return 1.0f / (1.0f + expf(-x)); }
__device__ __forceinline__ float gelu_exact(float x) {
    return 0.5f * x * (1.0f + erff(x * 0.70710678118654752f));
}

// Async copy of one 16-byte chunk global -> LDS (gfx1250 GLOBAL_LOAD_ASYNC_TO_LDS_B128,
// GVS mode: mem = SGPR base + 32-bit per-lane offset). Tracked by ASYNCcnt.
__device__ __forceinline__ void async_copy_b128(const void* base, unsigned gbyteoff,
                                                void* lds_ptr)
{
    unsigned lds = (unsigned)(uintptr_t)lds_ptr;   // flat LDS addr truncates to LDS offset
    asm volatile("global_load_async_to_lds_b128 %0, %1, %2"
                 :: "v"(lds), "v"(gbyteoff), "s"(base) : "memory");
}
__device__ __forceinline__ void wait_async0()
{
    asm volatile("s_wait_asynccnt 0x0" ::: "memory");
}

// ---------------------------------------------------------------------------
// Selector: per pathway, mean-pool batch 0 over S, sigmoid(mean @ sel_w + b),
// blend with priority, top-K(=4) indices (stable, descending).  grid=(2)
// ---------------------------------------------------------------------------
__global__ __launch_bounds__(256) void selector_topk_kernel(
    const float* cort, const float* brain,
    const float* c_w, const float* c_b,
    const float* b_w, const float* b_b, int* top)
{
    const float* x; const float* w; const float* bias; int* o;
    if (blockIdx.x == 0) { x = cort;  w = c_w; bias = c_b; o = top; }
    else                 { x = brain; w = b_w; bias = b_b; o = top + KSEL; }

    __shared__ float smean[DD];
    __shared__ float adj[NBLK];
    const int tid = threadIdx.x;

    for (int d = tid; d < DD; d += 256) {
        float s = 0.0f;
        for (int ss = 0; ss < SS; ++ss) s += x[(size_t)ss * DD + d];
        smean[d] = s / (float)SS;
    }
    __syncthreads();

    if (tid < NBLK) {
        float s = 0.0f;
        for (int d = 0; d < DD; ++d) s += smean[d] * w[d * NBLK + tid];
        s += bias[tid];
        adj[tid] = sigf(s) * 0.6f + 0.5f * 0.4f;
    }
    __syncthreads();

    if (tid == 0) {
        bool used[NBLK] = {};
        for (int k = 0; k < KSEL; ++k) {
            int best = 0; float bv = -1e30f;
            for (int j = 0; j < NBLK; ++j)
                if (!used[j] && adj[j] > bv) { bv = adj[j]; best = j; }
            used[best] = true;
            o[k] = best;
        }
    }
}

// ---------------------------------------------------------------------------
// pathway signal: out[b*D+d] = FUSION_STRENGTH * mean_s in[b,s,d]
// ---------------------------------------------------------------------------
__global__ __launch_bounds__(256) void signal_mean_kernel(const float* in, float* out)
{
    int idx = blockIdx.x * 256 + threadIdx.x;
    if (idx >= BB * DD) return;
    int b = idx / DD, d = idx % DD;
    const float* p = in + (size_t)b * SS * DD + d;
    float s = 0.0f;
    for (int i = 0; i < SS; ++i) s += p[(size_t)i * DD];
    out[idx] = s * (0.3f / (float)SS);
}

// ---------------------------------------------------------------------------
// LayerNorm over last dim (D=1024), bf16 output (feeds WMMA A directly).
// ---------------------------------------------------------------------------
__global__ __launch_bounds__(256) void layernorm_bf16_kernel(
    const float* x, const float* g_base, const float* b_base,
    const int* top, int step, __bf16* out)
{
    const int row = blockIdx.x;
    const int tid = threadIdx.x;
    const int sel = top[step];
    const float* g  = g_base + (size_t)sel * DD;
    const float* bb = b_base + (size_t)sel * DD;
    const float* xr = x + (size_t)row * DD;

    float v[4], lsum = 0.0f, lsq = 0.0f;
#pragma unroll
    for (int i = 0; i < 4; ++i) {
        v[i] = xr[tid + i * 256];
        lsum += v[i];
        lsq  += v[i] * v[i];
    }
    __shared__ float s1[256], s2[256];
    s1[tid] = lsum; s2[tid] = lsq;
    __syncthreads();
    for (int off = 128; off > 0; off >>= 1) {
        if (tid < off) { s1[tid] += s1[tid + off]; s2[tid] += s2[tid + off]; }
        __syncthreads();
    }
    const float mean = s1[0] * (1.0f / DD);
    const float var  = s2[0] * (1.0f / DD) - mean * mean;
    const float rstd = rsqrtf(var + 1e-5f);

    __bf16* orow = out + (size_t)row * DD;
#pragma unroll
    for (int i = 0; i < 4; ++i) {
        int d = tid + i * 256;
        orow[d] = (__bf16)((v[i] - mean) * rstd * g[d] + bb[d]);
    }
}

// ---------------------------------------------------------------------------
// fp32 -> bf16 elementwise (for the cross-projection inputs)
// ---------------------------------------------------------------------------
__global__ __launch_bounds__(256) void f32_to_bf16_kernel(const float* in, __bf16* out, long long n)
{
    long long i = (long long)blockIdx.x * 256 + threadIdx.x;
    if (i < n) out[i] = (__bf16)in[i];
}

// ---------------------------------------------------------------------------
// Weight transpose + convert: W fp32 [K][N] (block sel via top[step]) ->
// Wt bf16 [N][K].  32x32 LDS tile, coalesced both sides.  block=(32,8)
// ---------------------------------------------------------------------------
__global__ __launch_bounds__(256) void transpose_w_bf16_kernel(
    const float* Wbase, long long wStride, const int* top, int step,
    __bf16* Wt, int N, int Kd)
{
    const int sel = (top != nullptr) ? top[step] : 0;
    const float* W = Wbase + (long long)sel * wStride;

    __shared__ float t[32][33];
    const int n0 = blockIdx.x * 32;
    const int k0 = blockIdx.y * 32;
    const int x = threadIdx.x;   // 0..31
    const int y = threadIdx.y;   // 0..7

#pragma unroll
    for (int j = 0; j < 32; j += 8)
        t[y + j][x] = W[(size_t)(k0 + y + j) * N + n0 + x];
    __syncthreads();
#pragma unroll
    for (int j = 0; j < 32; j += 8)
        Wt[(size_t)(n0 + y + j) * Kd + k0 + x] = (__bf16)t[x][y + j];
}

// ---------------------------------------------------------------------------
// WMMA GEMM: C = A(bf16, MxK) @ Wt(bf16, [N][K] k-major) with fused epilogues.
//   MODE 0: attn   : out = res + 0.5*((acc+bias)*(1+sig(tg)*torsion))   (f32)
//   MODE 1: ff1    : out = gelu(acc+bias)                               (bf16)
//   MODE 2: ff1    : out = tanh(acc+bias)                               (bf16)
//   MODE 3: ff2    : out = res + 0.5*(acc+bias+signal[b]*sig(tg))       (f32)
//   MODE 4: crossA : out = acc                                          (f32)
//   MODE 5: crossB : out = out + acc + bias                             (f32)
// 256 threads = 8 waves; block tile 128(M) x 64(N) x 32(K).
// Wave (wm=w&3, wn=w>>2) owns rows wm*32..+32, cols wn*32..+32: 2x2 WMMA tiles.
// Tiles staged with gfx1250 async b128 copies (ASYNCcnt), double-buffered so
// the tile-(k+1) global->LDS DMA overlaps tile-k WMMA compute.
// ---------------------------------------------------------------------------
#define LDA 40   /* padded row stride (elements); 80B keeps 16B alignment */
#define LDB 40

template <int MODE>
__global__ __launch_bounds__(256) void gemm_wmma_kernel(
    const __bf16* A, const __bf16* Wt,
    const float* biasBase, long long bStride,
    const int* top, int step,
    const float* tgBase, const float* torsion,
    const float* res, const float* signal,
    float* outF, __bf16* outBf,
    int M, int N, int Kd)
{
    const int tid  = threadIdx.x;
    const int lane = tid & 31;
    const int wave = tid >> 5;
    const int wm   = wave & 3;   // 0..3 -> 32-row slab
    const int wn   = wave >> 2;  // 0..1 -> 32-col slab
    const int n0   = blockIdx.x * 64;
    const int m0   = blockIdx.y * 128;

    const int sel = (top != nullptr) ? top[step] : 0;
    const float* bias = biasBase ? (biasBase + (long long)sel * bStride) : nullptr;
    const float* tg   = tgBase ? (tgBase + (long long)sel * DD) : nullptr;

    __shared__ __bf16 sA[2][128 * LDA];   // [m][k] k-contiguous, 16B chunks
    __shared__ __bf16 sB[2][64 * LDB];    // [n][k] k-contiguous

    v8f acc00 = {}, acc01 = {}, acc10 = {}, acc11 = {};

    // per-thread staging chunks (16B each): A = 2 chunks, B = 1 chunk
    const int ar0 = tid >> 2,          ac0 = (tid & 3) * 8;
    const int ar1 = (tid + 256) >> 2,  ac1 = ((tid + 256) & 3) * 8;
    const int br  = tid >> 2,          bc  = (tid & 3) * 8;

    auto stage = [&](int kk, int buf) {
        __bf16* dA = &sA[buf][0];
        __bf16* dB = &sB[buf][0];
        async_copy_b128(A,  (unsigned)(((size_t)(m0 + ar0) * Kd + kk + ac0) * 2),
                        dA + ar0 * LDA + ac0);
        async_copy_b128(A,  (unsigned)(((size_t)(m0 + ar1) * Kd + kk + ac1) * 2),
                        dA + ar1 * LDA + ac1);
        async_copy_b128(Wt, (unsigned)(((size_t)(n0 + br) * Kd + kk + bc) * 2),
                        dB + br * LDB + bc);
    };

    // prologue: stage tile 0
    stage(0, 0);
    wait_async0();
    __syncthreads();

    int cur = 0;
    for (int k0 = 0; k0 < Kd; k0 += 32) {
        const int nxt = cur ^ 1;
        if (k0 + 32 < Kd) stage(k0 + 32, nxt);   // DMA next tile during compute

        // A fragments (16x32): row = lane&15; lanes 0-15 hold K 0-15,
        // lanes 16-31 hold K 16-31 (2 bf16 per VGPR).
        const int hh = lane >> 4;
        const int rr = lane & 15;
        v16bf a0, a1;
        const __bf16* ap0 = &sA[cur][(wm * 32 + rr) * LDA];
        const __bf16* ap1 = &sA[cur][(wm * 32 + 16 + rr) * LDA];
#pragma unroll
        for (int v = 0; v < 8; ++v) {
            int kk = 2 * v + hh * 16;
            a0[2 * v]     = ap0[kk];
            a0[2 * v + 1] = ap0[kk + 1];
            a1[2 * v]     = ap1[kk];
            a1[2 * v + 1] = ap1[kk + 1];
        }

        // B fragments (32x16): col = lane&15; lanes 0-15 K 0-15, lanes 16-31 K 16-31.
        const int col = lane & 15;
        const int kh  = (lane >> 4) * 16;
        v16bf b0, b1;
        const __bf16* bp0 = &sB[cur][(wn * 32 + col) * LDB + kh];
        const __bf16* bp1 = &sB[cur][(wn * 32 + 16 + col) * LDB + kh];
#pragma unroll
        for (int j = 0; j < 16; ++j) { b0[j] = bp0[j]; b1[j] = bp1[j]; }

        acc00 = __builtin_amdgcn_wmma_f32_16x16x32_bf16(false, a0, false, b0,
                                                        (short)0, acc00, false, false);
        acc01 = __builtin_amdgcn_wmma_f32_16x16x32_bf16(false, a0, false, b1,
                                                        (short)0, acc01, false, false);
        acc10 = __builtin_amdgcn_wmma_f32_16x16x32_bf16(false, a1, false, b0,
                                                        (short)0, acc10, false, false);
        acc11 = __builtin_amdgcn_wmma_f32_16x16x32_bf16(false, a1, false, b1,
                                                        (short)0, acc11, false, false);

        wait_async0();      // this wave's prefetch of tile k+1 landed
        __syncthreads();    // all waves done reading buffer `cur`
        cur = nxt;
    }

    // Epilogue.  C/D layout: VGPR r, lane n -> M = r + (n>>4)*8, N = n&15.
    const int lane16 = lane & 15;
    const int laneh  = lane >> 4;
#pragma unroll
    for (int fm = 0; fm < 2; ++fm) {
#pragma unroll
        for (int fn = 0; fn < 2; ++fn) {
            const v8f acc = (fm == 0) ? ((fn == 0) ? acc00 : acc01)
                                      : ((fn == 0) ? acc10 : acc11);
            const int gn = n0 + wn * 32 + fn * 16 + lane16;
#pragma unroll
            for (int r = 0; r < 8; ++r) {
                const int gm = m0 + wm * 32 + fm * 16 + laneh * 8 + r;
                const float val = acc[r];
                const size_t oi = (size_t)gm * N + gn;
                if (MODE == 0) {
                    float v = val + bias[gn];
                    v *= (1.0f + sigf(tg[gn]) * torsion[gn]);
                    outF[oi] = res[oi] + 0.5f * v;
                } else if (MODE == 1) {
                    outBf[oi] = (__bf16)gelu_exact(val + bias[gn]);
                } else if (MODE == 2) {
                    outBf[oi] = (__bf16)tanhf(val + bias[gn]);
                } else if (MODE == 3) {
                    float v = val + bias[gn] + signal[(size_t)(gm / SS) * DD + gn] * sigf(tg[gn]);
                    outF[oi] = res[oi] + 0.5f * v;
                } else if (MODE == 4) {
                    outF[oi] = val;
                } else { // MODE == 5
                    outF[oi] = outF[oi] + val + bias[gn];
                }
            }
        }
    }
}

// ---------------------------------------------------------------------------
extern "C" void kernel_launch(void* const* d_in, const int* in_sizes, int n_in,
                              void* d_out, int out_size, void* d_ws, size_t ws_size,
                              hipStream_t stream)
{
    (void)in_sizes; (void)n_in; (void)out_size; (void)ws_size;

    const float* cort_in  = (const float*)d_in[0];
    const float* brain_in = (const float*)d_in[1];
    const float* torsion  = (const float*)d_in[2];
    const float* c_ln1_g  = (const float*)d_in[3];
    const float* c_ln1_b  = (const float*)d_in[4];
    const float* c_attn_w = (const float*)d_in[5];
    const float* c_attn_b = (const float*)d_in[6];
    const float* c_ln2_g  = (const float*)d_in[7];
    const float* c_ln2_b  = (const float*)d_in[8];
    const float* c_ff1_w  = (const float*)d_in[9];
    const float* c_ff1_b  = (const float*)d_in[10];
    const float* c_ff2_w  = (const float*)d_in[11];
    const float* c_ff2_b  = (const float*)d_in[12];
    const float* c_tg     = (const float*)d_in[13];
    const float* b_ln1_g  = (const float*)d_in[14];
    const float* b_ln1_b  = (const float*)d_in[15];
    const float* b_attn_w = (const float*)d_in[16];
    const float* b_attn_b = (const float*)d_in[17];
    const float* b_ln2_g  = (const float*)d_in[18];
    const float* b_ln2_b  = (const float*)d_in[19];
    const float* b_ff1_w  = (const float*)d_in[20];
    const float* b_ff1_b  = (const float*)d_in[21];
    const float* b_ff2_w  = (const float*)d_in[22];
    const float* b_ff2_b  = (const float*)d_in[23];
    const float* b_tg     = (const float*)d_in[24];
    const float* c_sel_w  = (const float*)d_in[25];
    const float* c_sel_b  = (const float*)d_in[26];
    const float* b_sel_w  = (const float*)d_in[27];
    const float* b_sel_b  = (const float*)d_in[28];
    const float* cross_w  = (const float*)d_in[29];
    const float* cross_b  = (const float*)d_in[30];

    // output regions: [cortical_h | brainstem_h | fused], each B*S*D fp32
    float* cort_out  = (float*)d_out;
    float* brain_out = cort_out + BSD;
    float* fused_out = cort_out + 2 * BSD;

    // workspace layout
    char*   ws    = (char*)d_ws;
    int*    top   = (int*)ws;                                   // 2*KSEL ints
    float*  c_sig = (float*)(ws + 256);                         // B*D
    float*  b_sig = c_sig + (size_t)BB * DD;                    // B*D
    float*  h1    = (float*)(ws + 256 + 2 * (size_t)BB * DD * sizeof(float));
    __bf16* Xbf   = (__bf16*)((char*)h1 + BSD * sizeof(float)); // B*S*D bf16
    __bf16* mid   = Xbf + BSD;                                  // B*S*2D bf16
    __bf16* Wt    = mid + 2 * BSD;                              // <= 2048x1024 bf16

    const long long DxD  = (long long)DD * DD;
    const long long Dx2D = (long long)DD * 2 * DD;

    // 1. routing + fusion signals (independent)
    selector_topk_kernel<<<2, 256, 0, stream>>>(cort_in, brain_in,
                                                c_sel_w, c_sel_b, b_sel_w, b_sel_b, top);
    signal_mean_kernel<<<(BB * DD + 255) / 256, 256, 0, stream>>>(brain_in, b_sig);
    signal_mean_kernel<<<(BB * DD + 255) / 256, 256, 0, stream>>>(cort_in, c_sig);

    const dim3 tb(32, 8);
    const dim3 tDD(DD / 32, DD / 32);          // transpose 1024x1024
    const dim3 tD2D(2 * DD / 32, DD / 32);     // W [K=1024][N=2048]
    const dim3 t2DD(DD / 32, 2 * DD / 32);     // W [K=2048][N=1024]
    const dim3 gN1(DD / 64, MM / 128);         // GEMM N=1024
    const dim3 gN2(2 * DD / 64, MM / 128);     // GEMM N=2048

    // 2. cortical pathway (4 selected blocks, sequential; GELU, FF=2D)
    for (int step = 0; step < KSEL; ++step) {
        const float* h = (step == 0) ? cort_in : cort_out;
        layernorm_bf16_kernel<<<MM, 256, 0, stream>>>(h, c_ln1_g, c_ln1_b, top, step, Xbf);
        transpose_w_bf16_kernel<<<tDD, tb, 0, stream>>>(c_attn_w, DxD, top, step, Wt, DD, DD);
        gemm_wmma_kernel<0><<<gN1, 256, 0, stream>>>(Xbf, Wt, c_attn_b, DD,
            top, step, c_tg, torsion, h, nullptr, h1, nullptr, MM, DD, DD);
        layernorm_bf16_kernel<<<MM, 256, 0, stream>>>(h1, c_ln2_g, c_ln2_b, top, step, Xbf);
        transpose_w_bf16_kernel<<<tD2D, tb, 0, stream>>>(c_ff1_w, Dx2D, top, step, Wt, 2 * DD, DD);
        gemm_wmma_kernel<1><<<gN2, 256, 0, stream>>>(Xbf, Wt, c_ff1_b, 2 * DD,
            top, step, nullptr, nullptr, nullptr, nullptr, nullptr, mid, MM, 2 * DD, DD);
        transpose_w_bf16_kernel<<<t2DD, tb, 0, stream>>>(c_ff2_w, Dx2D, top, step, Wt, DD, 2 * DD);
        gemm_wmma_kernel<3><<<gN1, 256, 0, stream>>>(mid, Wt, c_ff2_b, DD,
            top, step, c_tg, nullptr, h1, b_sig, cort_out, nullptr, MM, DD, 2 * DD);
    }

    // 3. brainstem pathway (tanh, FF=D)
    for (int step = 0; step < KSEL; ++step) {
        const float* h = (step == 0) ? brain_in : brain_out;
        layernorm_bf16_kernel<<<MM, 256, 0, stream>>>(h, b_ln1_g, b_ln1_b, top + KSEL, step, Xbf);
        transpose_w_bf16_kernel<<<tDD, tb, 0, stream>>>(b_attn_w, DxD, top + KSEL, step, Wt, DD, DD);
        gemm_wmma_kernel<0><<<gN1, 256, 0, stream>>>(Xbf, Wt, b_attn_b, DD,
            top + KSEL, step, b_tg, torsion, h, nullptr, h1, nullptr, MM, DD, DD);
        layernorm_bf16_kernel<<<MM, 256, 0, stream>>>(h1, b_ln2_g, b_ln2_b, top + KSEL, step, Xbf);
        transpose_w_bf16_kernel<<<tDD, tb, 0, stream>>>(b_ff1_w, DxD, top + KSEL, step, Wt, DD, DD);
        gemm_wmma_kernel<2><<<gN1, 256, 0, stream>>>(Xbf, Wt, b_ff1_b, DD,
            top + KSEL, step, nullptr, nullptr, nullptr, nullptr, nullptr, mid, MM, DD, DD);
        transpose_w_bf16_kernel<<<tDD, tb, 0, stream>>>(b_ff2_w, DxD, top + KSEL, step, Wt, DD, DD);
        gemm_wmma_kernel<3><<<gN1, 256, 0, stream>>>(mid, Wt, b_ff2_b, DD,
            top + KSEL, step, b_tg, nullptr, h1, c_sig, brain_out, nullptr, MM, DD, DD);
    }

    // 4. cross projection: fused = cort_h @ W[:D] + brain_h @ W[D:] + bias
    const int cvtBlocks = (int)((BSD + 255) / 256);
    f32_to_bf16_kernel<<<cvtBlocks, 256, 0, stream>>>(cort_out, Xbf, (long long)BSD);
    transpose_w_bf16_kernel<<<tDD, tb, 0, stream>>>(cross_w, 0, nullptr, 0, Wt, DD, DD);
    gemm_wmma_kernel<4><<<gN1, 256, 0, stream>>>(Xbf, Wt, nullptr, 0,
        nullptr, 0, nullptr, nullptr, nullptr, nullptr, fused_out, nullptr, MM, DD, DD);
    f32_to_bf16_kernel<<<cvtBlocks, 256, 0, stream>>>(brain_out, Xbf, (long long)BSD);
    transpose_w_bf16_kernel<<<tDD, tb, 0, stream>>>(cross_w + DxD, 0, nullptr, 0, Wt, DD, DD);
    gemm_wmma_kernel<5><<<gN1, 256, 0, stream>>>(Xbf, Wt, cross_b, 0,
        nullptr, 0, nullptr, nullptr, nullptr, nullptr, fused_out, nullptr, MM, DD, DD);
}
